// Spa_RSA_42528766165339
// MI455X (gfx1250) — matrix operand order repaired and checked by reading, not compiled, not run on hardware
//
#include <hip/hip_runtime.h>

// MaxViT block+grid attention for MI455X (gfx1250, wave32, WMMA).
// All matmuls via v_wmma_f32_16x16x32_bf16; intermediates bf16.
// Weights pre-transposed/converted to bf16 N-major once per launch so every
// B fragment (global or LDS) is read as 16B runs per lane.

typedef __attribute__((ext_vector_type(16))) __bf16 v16bf;
typedef __attribute__((ext_vector_type(8)))  __bf16 v8bf;
typedef __attribute__((ext_vector_type(8)))  float  v8f;

#define B_    4
#define IMG   128
#define DIM   96
#define INNER 384
#define HH    4
#define DH    48
#define NWIN  256
#define MM    64
#define SEQ2  256
#define QKVC  1152
#define SCALE 0.14433756729740643f

union Frag { v16bf v; v8bf h[2]; __bf16 e[16]; };

// A 16x32 (ISA 7.12.2): per lane two contiguous 16B chunks (k = half*8 + 0..7, +16)
__device__ __forceinline__ v16bf load_A32(const __bf16* base, int ld, int lane) {
  const int half = (lane >> 4) & 1, r = lane & 15;
  const __bf16* p = base + r * ld + half * 8;
  Frag a;
  a.h[0] = *(const v8bf*)(p);
  a.h[1] = *(const v8bf*)(p + 16);
  return a.v;
}
// A with 16 valid K (tail for DH=48): chunk2 zeroed, no divergence
__device__ __forceinline__ v16bf load_A16(const __bf16* base, int ld, int lane) {
  const int half = (lane >> 4) & 1, r = lane & 15;
  Frag a;
  a.h[0] = *(const v8bf*)(base + r * ld + half * 8);
  v8bf z = {};
  a.h[1] = z;
  return a.v;
}
// Transposed B 32x16: B[k][n] = base[n*ld + k]; per lane two 16B chunks (k = half*8 + 0..7, +16)
__device__ __forceinline__ v16bf load_Bt32(const __bf16* base, int ld, int lane) {
  const int half = (lane >> 4) & 1, n = lane & 15;
  const __bf16* p = base + n * ld + half * 8;
  Frag b;
  b.h[0] = *(const v8bf*)(p);
  b.h[1] = *(const v8bf*)(p + 16);
  return b.v;
}
// Transposed B with 16 valid K: lanes load k = half*8 + 0..7, chunk2 zeroed
__device__ __forceinline__ v16bf load_Bt16(const __bf16* base, int ld, int lane) {
  const int half = (lane >> 4) & 1, n = lane & 15;
  Frag b;
  v8bf z = {};
  b.h[0] = *(const v8bf*)(base + n * ld + half * 8);
  b.h[1] = z;
  return b.v;
}

__device__ __forceinline__ v8f wmma_bf16(v16bf a, v16bf b, v8f c) {
  return __builtin_amdgcn_wmma_f32_16x16x32_bf16(false, a, false, b, (short)0, c, false, false);
}

// ---------------------------------------------------------------------------
// Kernel 0: one-time weight transpose + f32->bf16 convert.
// WT  [1152][96]: WT[n][k]  = (n<384 ? Wq[k][n] : Wkv[k][n-384])
// WoT [96][384] : WoT[n][k] = Wo[k][n]
// ---------------------------------------------------------------------------
__global__ __launch_bounds__(256) void wconv_kernel(const float* __restrict__ Wq,
                                                    const float* __restrict__ Wkv,
                                                    const float* __restrict__ Wo,
                                                    __bf16* __restrict__ WT,
                                                    __bf16* __restrict__ WoT) {
  const int i = blockIdx.x * 256 + threadIdx.x;
  if (i < QKVC * DIM) {
    const int n = i / DIM, k = i % DIM;
    const float v = (n < INNER) ? Wq[(size_t)k * INNER + n]
                                : Wkv[(size_t)k * (2 * INNER) + (n - INNER)];
    WT[i] = (__bf16)v;
  } else if (i < QKVC * DIM + DIM * INNER) {
    const int j = i - QKVC * DIM;
    const int n = j / INNER, k = j % INNER;
    WoT[j] = (__bf16)Wo[(size_t)k * DIM + n];
  }
}

// ---------------------------------------------------------------------------
// Kernel 1: QKV projection -> window-partitioned bf16 [b, win, pos, 1152]
// ---------------------------------------------------------------------------
__global__ __launch_bounds__(256) void qkv_kernel(const float* __restrict__ x,
                                                  const __bf16* __restrict__ WT,
                                                  __bf16* __restrict__ qkv) {
  const int blk = blockIdx.x;
  const int bb = blk >> 8, win = blk & 255;
  const int tid = threadIdx.x, lane = tid & 31, wv = tid >> 5;

  __shared__ __attribute__((aligned(16))) __bf16 xs[MM][DIM + 8];

  for (int t = tid; t < MM * DIM; t += 256) {
    const int pos = t / DIM, c = t % DIM;
    const int y  = ((win >> 4) << 3) + (pos >> 3);
    const int xc = ((win & 15) << 3) + (pos & 7);
    xs[pos][c] = (__bf16)x[((size_t)(bb * IMG + y) * IMG + xc) * DIM + c];
  }
  __syncthreads();

  v16bf afr[4][3];
#pragma unroll
  for (int mt = 0; mt < 4; ++mt)
#pragma unroll
    for (int ks = 0; ks < 3; ++ks)
      afr[mt][ks] = load_A32(&xs[mt * 16][ks * 32], DIM + 8, lane);

  __bf16* outb = qkv + (size_t)blk * MM * QKVC;
  for (int nt = wv; nt < 72; nt += 8) {
    const int n0 = nt * 16;
    v8f acc[4] = {};
#pragma unroll
    for (int ks = 0; ks < 3; ++ks) {
      const v16bf bf = load_Bt32(WT + (size_t)n0 * DIM + ks * 32, DIM, lane);
#pragma unroll
      for (int mt = 0; mt < 4; ++mt) acc[mt] = wmma_bf16(afr[mt][ks], bf, acc[mt]);
    }
    const int half = (lane >> 4) & 1, col = lane & 15;
#pragma unroll
    for (int mt = 0; mt < 4; ++mt)
#pragma unroll
      for (int r = 0; r < 8; ++r) {
        const int row = mt * 16 + r + (half ? 8 : 0);
        outb[(size_t)row * QKVC + n0 + col] = (__bf16)acc[mt][r];
      }
  }
}

// ---------------------------------------------------------------------------
// Kernel 2: local window attention, one block per (b, win, head)
// ---------------------------------------------------------------------------
__global__ __launch_bounds__(256) void local_attn_kernel(const __bf16* __restrict__ qkv,
                                                         const float* __restrict__ pos1,
                                                         __bf16* __restrict__ cat) {
  const int blk = blockIdx.x;
  const int h = blk & 3;
  const int bw = blk >> 2;
  const int tid = threadIdx.x, lane = tid & 31, wv = tid >> 5;

  __shared__ __attribute__((aligned(16))) __bf16 qs[MM][DH + 8];
  __shared__ __attribute__((aligned(16))) __bf16 kk[MM][DH + 8];
  __shared__ __attribute__((aligned(16))) __bf16 vT[DH][MM + 8];   // V transposed
  __shared__ __attribute__((aligned(16))) __bf16 P[MM][MM + 8];
  __shared__ float S[MM][MM + 1];
  __shared__ float red[MM][4];

  const __bf16* base = qkv + (size_t)bw * MM * QKVC;
  for (int t = tid; t < MM * DH; t += 256) {
    const int r = t / DH, c = t % DH;
    qs[r][c] = base[(size_t)r * QKVC + h * DH + c];
    kk[r][c] = base[(size_t)r * QKVC + INNER + h * DH + c];
    vT[c][r] = base[(size_t)r * QKVC + 2 * INNER + h * DH + c];
  }
  __syncthreads();

  const float* pb = pos1 + (size_t)h * MM * MM;
  for (int t = wv; t < 16; t += 8) {
    const int mt = t >> 2, nt = t & 3;
    v8f acc = {};
    acc = wmma_bf16(load_A32(&qs[mt * 16][0],  DH + 8, lane),
                    load_Bt32(&kk[nt * 16][0], DH + 8, lane), acc);
    acc = wmma_bf16(load_A16(&qs[mt * 16][32],  DH + 8, lane),
                    load_Bt16(&kk[nt * 16][32], DH + 8, lane), acc);
    const int half = (lane >> 4) & 1, col = lane & 15;
#pragma unroll
    for (int r = 0; r < 8; ++r) {
      const int row = mt * 16 + r + (half ? 8 : 0);
      const int cc = nt * 16 + col;
      S[row][cc] = acc[r] * SCALE + pb[row * MM + cc];
    }
  }
  __syncthreads();

  {
    const int row = tid & 63, q = tid >> 6;
    float m = -3.0e38f;
    for (int c = q * 16; c < q * 16 + 16; ++c) m = fmaxf(m, S[row][c]);
    red[row][q] = m;
    __syncthreads();
    m = fmaxf(fmaxf(red[row][0], red[row][1]), fmaxf(red[row][2], red[row][3]));
    float s = 0.f;
    for (int c = q * 16; c < q * 16 + 16; ++c) {
      float e = __expf(S[row][c] - m); S[row][c] = e; s += e;
    }
    __syncthreads();
    red[row][q] = s;
    __syncthreads();
    const float inv = 1.f / (red[row][0] + red[row][1] + red[row][2] + red[row][3]);
    for (int c = q * 16; c < q * 16 + 16; ++c) P[row][c] = (__bf16)(S[row][c] * inv);
  }
  __syncthreads();

  __bf16* ob = cat + (size_t)bw * MM * INNER + h * DH;
  for (int t = wv; t < 12; t += 8) {
    const int mt = t / 3, nt = t % 3;
    v8f acc = {};
#pragma unroll
    for (int ks = 0; ks < 2; ++ks)
      acc = wmma_bf16(load_A32(&P[mt * 16][ks * 32], MM + 8, lane),
                      load_Bt32(&vT[nt * 16][ks * 32], MM + 8, lane), acc);
    const int half = (lane >> 4) & 1, col = lane & 15;
#pragma unroll
    for (int r = 0; r < 8; ++r) {
      const int row = mt * 16 + r + (half ? 8 : 0);
      ob[(size_t)row * INNER + nt * 16 + col] = (__bf16)acc[r];
    }
  }
}

// ---------------------------------------------------------------------------
// Kernel 3: grid attention, one block per (b, pos, head)
// ---------------------------------------------------------------------------
__global__ __launch_bounds__(256) void grid_attn_kernel(const __bf16* __restrict__ qkv,
                                                        const float* __restrict__ pos2,
                                                        __bf16* __restrict__ cat) {
  const int blk = blockIdx.x;
  const int h = blk & 3;
  const int bp = blk >> 2;
  const int bb = bp >> 6, pos = bp & 63;
  const int tid = threadIdx.x, lane = tid & 31, wv = tid >> 5;

  extern __shared__ __attribute__((aligned(16))) char smem[];
  __bf16* kk  = (__bf16*)smem;            // [256][56]
  __bf16* vT  = kk + 256 * 56;            // [48][264] (V transposed)
  __bf16* qs  = vT + 48 * 264;            // [64][56]
  float*  S   = (float*)(qs + 64 * 56);   // [64][260]
  __bf16* P   = (__bf16*)(S + 64 * 260);  // [64][264]
  float*  red = (float*)(P + 64 * 264);   // [64][4]

  const __bf16* base = qkv + ((size_t)bb * NWIN * MM + pos) * QKVC;
  for (int t = tid; t < SEQ2 * DH; t += 256) {
    const int j = t / DH, c = t % DH;
    const __bf16* rp = base + (size_t)j * MM * QKVC;
    kk[j * 56 + c]  = rp[INNER + h * DH + c];
    vT[c * 264 + j] = rp[2 * INNER + h * DH + c];
  }
  __syncthreads();

  const float* pb = pos2 + (size_t)h * SEQ2 * SEQ2;
  __bf16* ob = cat + (size_t)bb * NWIN * MM * INNER + pos * INNER + INNER / 2 + h * DH;

  for (int rb = 0; rb < 4; ++rb) {
    for (int t = tid; t < 64 * DH; t += 256) {
      const int j = t / DH, c = t % DH;
      qs[j * 56 + c] = base[(size_t)(rb * 64 + j) * MM * QKVC + h * DH + c];
    }
    __syncthreads();

    for (int t = wv; t < 64; t += 8) {
      const int mt = t >> 4, nt = t & 15;
      v8f acc = {};
      acc = wmma_bf16(load_A32(qs + mt * 16 * 56, 56, lane),
                      load_Bt32(kk + nt * 16 * 56, 56, lane), acc);
      acc = wmma_bf16(load_A16(qs + mt * 16 * 56 + 32, 56, lane),
                      load_Bt16(kk + nt * 16 * 56 + 32, 56, lane), acc);
      const int half = (lane >> 4) & 1, col = lane & 15;
#pragma unroll
      for (int r = 0; r < 8; ++r) {
        const int row = mt * 16 + r + (half ? 8 : 0);
        const int cc = nt * 16 + col;
        S[row * 260 + cc] = acc[r] * SCALE + pb[(size_t)(rb * 64 + row) * SEQ2 + cc];
      }
    }
    __syncthreads();

    {
      const int row = tid & 63, q = tid >> 6;
      float m = -3.0e38f;
      for (int c = q * 64; c < q * 64 + 64; ++c) m = fmaxf(m, S[row * 260 + c]);
      red[row * 4 + q] = m;
      __syncthreads();
      m = fmaxf(fmaxf(red[row * 4], red[row * 4 + 1]),
                fmaxf(red[row * 4 + 2], red[row * 4 + 3]));
      float s = 0.f;
      for (int c = q * 64; c < q * 64 + 64; ++c) {
        float e = __expf(S[row * 260 + c] - m); S[row * 260 + c] = e; s += e;
      }
      __syncthreads();
      red[row * 4 + q] = s;
      __syncthreads();
      const float inv = 1.f / (red[row * 4] + red[row * 4 + 1] +
                               red[row * 4 + 2] + red[row * 4 + 3]);
      for (int c = q * 64; c < q * 64 + 64; ++c)
        P[row * 264 + c] = (__bf16)(S[row * 260 + c] * inv);
    }
    __syncthreads();

    for (int t = wv; t < 12; t += 8) {
      const int mt = t / 3, nt = t % 3;
      v8f acc = {};
#pragma unroll
      for (int ks = 0; ks < 8; ++ks)
        acc = wmma_bf16(load_A32(P + mt * 16 * 264 + ks * 32, 264, lane),
                        load_Bt32(vT + nt * 16 * 264 + ks * 32, 264, lane), acc);
      const int half = (lane >> 4) & 1, col = lane & 15;
#pragma unroll
      for (int r = 0; r < 8; ++r) {
        const int row = rb * 64 + mt * 16 + r + (half ? 8 : 0);
        ob[(size_t)row * MM * INNER + nt * 16 + col] = (__bf16)acc[r];
      }
    }
    __syncthreads();
  }
}

// ---------------------------------------------------------------------------
// Kernel 4: output projection + bias + window-merge scatter
// ---------------------------------------------------------------------------
__global__ __launch_bounds__(256) void out_proj_kernel(const __bf16* __restrict__ cat,
                                                       const __bf16* __restrict__ WoT,
                                                       const float* __restrict__ bo,
                                                       float* __restrict__ out) {
  const int blk = blockIdx.x;
  const int bb = blk >> 8, win = blk & 255;
  const int tid = threadIdx.x, lane = tid & 31, wv = tid >> 5;

  __shared__ __attribute__((aligned(16))) __bf16 As[MM][INNER + 8];

  const __bf16* cb = cat + (size_t)blk * MM * INNER;
  for (int t = tid; t < MM * INNER; t += 256) {
    const int r = t / INNER, c = t % INNER;
    As[r][c] = cb[(size_t)r * INNER + c];
  }
  __syncthreads();

  for (int t = wv; t < 24; t += 8) {
    const int mt = t / 6, nt = t % 6;
    v8f acc = {};
    for (int ks = 0; ks < 12; ++ks)
      acc = wmma_bf16(load_A32(&As[mt * 16][ks * 32], INNER + 8, lane),
                      load_Bt32(WoT + (size_t)(nt * 16) * INNER + ks * 32, INNER, lane), acc);
    const int half = (lane >> 4) & 1, col = lane & 15;
    const int cc = nt * 16 + col;
    const float bias = bo[cc];
#pragma unroll
    for (int r = 0; r < 8; ++r) {
      const int row = mt * 16 + r + (half ? 8 : 0);
      const int y  = ((win >> 4) << 3) + (row >> 3);
      const int xc = ((win & 15) << 3) + (row & 7);
      out[((size_t)(bb * IMG + y) * IMG + xc) * DIM + cc] = acc[r] + bias;
    }
  }
}

extern "C" void kernel_launch(void* const* d_in, const int* in_sizes, int n_in,
                              void* d_out, int out_size, void* d_ws, size_t ws_size,
                              hipStream_t stream) {
  const float* x    = (const float*)d_in[0];
  const float* Wq   = (const float*)d_in[1];
  const float* Wkv  = (const float*)d_in[2];
  const float* Wo   = (const float*)d_in[3];
  const float* bo   = (const float*)d_in[4];
  const float* pos1 = (const float*)d_in[5];
  const float* pos2 = (const float*)d_in[6];

  // workspace: qkv bf16 (151 MB) + concat bf16 (50 MB) + transposed weights (~288 KB)
  char* ws = (char*)d_ws;
  __bf16* qkv = (__bf16*)ws;
  ws += (size_t)B_ * NWIN * MM * QKVC * sizeof(__bf16);
  __bf16* cat = (__bf16*)ws;
  ws += (size_t)B_ * NWIN * MM * INNER * sizeof(__bf16);
  __bf16* WT  = (__bf16*)ws;                 // [1152][96]
  ws += (size_t)QKVC * DIM * sizeof(__bf16);
  __bf16* WoT = (__bf16*)ws;                 // [96][384]

  dim3 blk(256);
  const int wconv_elems = QKVC * DIM + DIM * INNER;
  wconv_kernel<<<dim3((wconv_elems + 255) / 256), blk, 0, stream>>>(Wq, Wkv, Wo, WT, WoT);

  qkv_kernel<<<dim3(B_ * NWIN), blk, 0, stream>>>(x, WT, qkv);
  local_attn_kernel<<<dim3(B_ * NWIN * HH), blk, 0, stream>>>(qkv, pos1, cat);

  const size_t smem3 = (size_t)(256 * 56 + 48 * 264 + 64 * 56) * sizeof(__bf16) +
                       (size_t)64 * 260 * sizeof(float) +
                       (size_t)64 * 264 * sizeof(__bf16) +
                       (size_t)64 * 4 * sizeof(float);   // ~159 KB (< 320 KB/WGP)
  grid_attn_kernel<<<dim3(B_ * MM * HH), blk, smem3, stream>>>(qkv, pos2, cat);

  out_proj_kernel<<<dim3(B_ * NWIN), blk, 0, stream>>>(cat, WoT, bo, (float*)d_out);
}
